// ResidualVectorQuantizerEMA_76897094468434
// MI455X (gfx1250) — matrix operand order repaired and checked by reading, not compiled, not run on hardware
//
#include <hip/hip_runtime.h>
#include <hip/hip_bf16.h>

// ---------------------------------------------------------------------------
// Residual VQ (4 layers, N=131072 tokens, D=256, K=1024 codes) for gfx1250.
// Matmul path: v_wmma_f32_16x16x32_bf16 (bf16 inputs, f32 accumulate).
// Codebook tiles staged via GLOBAL_LOAD_ASYNC_TO_LDS_B128 (ASYNCcnt),
// double-buffered so staging of tile t+1 overlaps the WMMAs of tile t.
// ---------------------------------------------------------------------------

#define N_TOKENS      131072
#define CODE_DIM      256
#define CB_SIZE       1024
#define NUM_LAYERS    4
#define TOK_PER_WG    128          // 8 waves * 16 tokens
#define WG_THREADS    256
#define NTILES        (CB_SIZE / 16)

#ifndef USE_ASYNC_STAGE
#define USE_ASYNC_STAGE 1          // set 0 to fall back to sync load+ds_store
#endif

typedef __attribute__((ext_vector_type(16))) __bf16 v16bf;
typedef __attribute__((ext_vector_type(8)))  __bf16 v8bf;
typedef __attribute__((ext_vector_type(8)))  float  v8f;

// ---------------------------------------------------------------------------
// Prep: f32 codebooks -> bf16 copy in ws, per-code squared norms, zero SSE.
// ---------------------------------------------------------------------------
__global__ __launch_bounds__(256)
void rvq_prep(const float* __restrict__ cb,
              __bf16* __restrict__ cb16,
              float* __restrict__ enorm,
              float* __restrict__ sse)
{
    int row = blockIdx.x * blockDim.x + threadIdx.x;
    if (row < NUM_LAYERS) sse[row] = 0.0f;
    if (row >= NUM_LAYERS * CB_SIZE) return;
    const float* r = cb + (size_t)row * CODE_DIM;
    float s = 0.0f;
    #pragma unroll 8
    for (int d = 0; d < CODE_DIM; ++d) {
        float v = r[d];
        s += v * v;
        cb16[(size_t)row * CODE_DIM + d] = (__bf16)v;
    }
    enorm[row] = s;
}

// ---------------------------------------------------------------------------
// Async stage: 16 code rows (16 x 256 bf16 = 8 KB) -> LDS buffer.
// 256 threads x 32 B each, issued as two async b128 transfers per thread.
// Tracked with ASYNCcnt; no VGPR data round-trip.
// ---------------------------------------------------------------------------
__device__ __forceinline__ void stage_tile_async(const __bf16* gsrc,
                                                 __bf16* ldst, int tid)
{
#if USE_ASYNC_STAGE
    const int off = tid * 32;                                // bytes
    unsigned long long ga = (unsigned long long)(uintptr_t)((const char*)gsrc + off);
    unsigned int       la = (unsigned int)(uintptr_t)((char*)ldst + off);
    asm volatile("global_load_async_to_lds_b128 %0, %1, off"
                 :: "v"(la), "v"(ga) : "memory");
    asm volatile("global_load_async_to_lds_b128 %0, %1, off offset:16"
                 :: "v"(la), "v"(ga) : "memory");
#else
    const int r  = tid >> 4;              // code row 0..15
    const int c0 = (tid & 15) * 16;       // dim offset
    const uint4* gs = (const uint4*)(gsrc + (size_t)r * CODE_DIM + c0);
    *(uint4*)(ldst + r * CODE_DIM + c0) = *gs;
#endif
}

__device__ __forceinline__ void wait_async_staging()
{
#if USE_ASYNC_STAGE
    asm volatile("s_wait_asynccnt 0x0" ::: "memory");
#endif
}

// ---------------------------------------------------------------------------
// One quantization layer. grid = N_TOKENS / TOK_PER_WG = 1024 blocks.
// ---------------------------------------------------------------------------
__global__ __launch_bounds__(WG_THREADS)
void rvq_layer(const float* __restrict__ z_flat,   // layer-0 residual source
               float* __restrict__ resid,          // ws residual [N,D] f32
               const __bf16* __restrict__ cb16,    // this layer [K,D] bf16
               const float*  __restrict__ cb32,    // this layer [K,D] f32
               const float*  __restrict__ enorm,   // this layer [K]
               float* __restrict__ qsum,           // d_out + 1, [N,D]
               float* __restrict__ idx_out,        // d_out + 1 + N*D + l*N
               float* __restrict__ sse_accum,      // ws slot for this layer
               int layer)
{
    const int tid  = threadIdx.x;
    const int wave = tid >> 5;          // 0..7
    const int lane = tid & 31;          // wave32
    const int laneLo = lane & 15;
    const int laneHi = lane >> 4;       // 0 or 1
    const int wg_tok0   = blockIdx.x * TOK_PER_WG;
    const int wave_tok0 = wg_tok0 + wave * 16;

    __shared__ __bf16 lds_tile[2][16 * CODE_DIM];   // 2 x 8 KB double buffer
    __shared__ int    lds_idx[TOK_PER_WG];
    __shared__ float  lds_sse;

    const float* src = (layer == 0) ? z_flat : resid;

    // ---- B matrix (tokens), resident in registers for all 64 code tiles.
    // B is 32x16 (KxN) per chunk: lane n (0-15) holds dims [32c .. 32c+15] of
    // token n; lane n+16 holds dims [32c+16 .. 32c+31].  (contiguous k order, matched by A)
    v16bf B[CODE_DIM / 32];
    {
        const float* brow = src + (size_t)(wave_tok0 + laneLo) * CODE_DIM;
        #pragma unroll
        for (int c = 0; c < CODE_DIM / 32; ++c) {
            const int base = 32 * c + laneHi * 16;
            const float4* p = (const float4*)(brow + base);
            float4 f0 = p[0], f1 = p[1], f2 = p[2], f3 = p[3];
            v16bf b;
            b[0]=(__bf16)f0.x;  b[1]=(__bf16)f0.y;  b[2]=(__bf16)f0.z;  b[3]=(__bf16)f0.w;
            b[4]=(__bf16)f1.x;  b[5]=(__bf16)f1.y;  b[6]=(__bf16)f1.z;  b[7]=(__bf16)f1.w;
            b[8]=(__bf16)f2.x;  b[9]=(__bf16)f2.y;  b[10]=(__bf16)f2.z; b[11]=(__bf16)f2.w;
            b[12]=(__bf16)f3.x; b[13]=(__bf16)f3.y; b[14]=(__bf16)f3.z; b[15]=(__bf16)f3.w;
            B[c] = b;
        }
    }

    // Running argmin.  With A = codes (M) and B = tokens (N), the f32 C tile
    // layout gives lane n (0-15) rows m=0..7 for token n in its 8 VGPRs and
    // lane n+16 rows m=8..15 for the same token.
    float bestv = 3.0e38f;
    int   besti = 0;

    // prologue: stage tile 0 into buffer 0
    stage_tile_async(cb16, &lds_tile[0][0], tid);

    for (int tile = 0; tile < NTILES; ++tile) {
        const int buf = tile & 1;
        wait_async_staging();    // this wave's stage of `buf` landed in LDS
        __syncthreads();         // all waves' stores visible; other buffer free
        if (tile + 1 < NTILES)   // overlap next stage with this tile's WMMAs
            stage_tile_async(cb16 + (size_t)(tile + 1) * 16 * CODE_DIM,
                             &lds_tile[buf ^ 1][0], tid);

        v8f acc = {0.f, 0.f, 0.f, 0.f, 0.f, 0.f, 0.f, 0.f};
        #pragma unroll
        for (int c = 0; c < CODE_DIM / 32; ++c) {
            // A per-lane (16-bit 16x32): lanes 0-15: M=lane, K = {32c+0..15};
            // lanes 16-31: M=lane-16, K = {32c+16..31}.
            const int k0 = 32 * c + laneHi * 16;
            v8bf lo = *(const v8bf*)(&lds_tile[buf][laneLo * CODE_DIM + k0]);
            v8bf hi = *(const v8bf*)(&lds_tile[buf][laneLo * CODE_DIM + k0 + 8]);
            v16bf A = __builtin_shufflevector(lo, hi,
                        0,1,2,3,4,5,6,7,8,9,10,11,12,13,14,15);
            acc = __builtin_amdgcn_wmma_f32_16x16x32_bf16(
                      false, A, false, B[c], (short)0, acc, false, false);
        }

        const int codebase = tile * 16 + laneHi * 8;
        #pragma unroll
        for (int r = 0; r < 8; ++r) {
            // dist ~ ||e||^2 - 2 z.e   (||z||^2 constant per token)
            float s = enorm[codebase + r] - 2.0f * acc[r];
            if (s < bestv) { bestv = s; besti = codebase + r; }
        }
    }

    // merge the two half-lanes holding the same token (lane ^ 16)
    {
        float ov = __shfl_xor(bestv, 16, 32);
        int   oi = __shfl_xor(besti, 16, 32);
        if (ov < bestv || (ov == bestv && oi < besti)) { bestv = ov; besti = oi; }
    }
    if (lane < 16) lds_idx[wave * 16 + lane] = besti;
    if (tid == 0)  lds_sse = 0.0f;
    __syncthreads();

    // ---- gather / residual update / loss, coalesced over 128x256 elements
    float sse = 0.0f;
    for (int e = tid; e < TOK_PER_WG * CODE_DIM; e += WG_THREADS) {
        const int t = e >> 8;               // token in block
        const int d = e & (CODE_DIM - 1);
        const size_t go = (size_t)(wg_tok0 + t) * CODE_DIM + d;
        const int code = lds_idx[t];
        const float r_old = src[go];
        const float q     = cb32[(size_t)code * CODE_DIM + d];
        const float diff  = q - r_old;
        sse += diff * diff;
        resid[go] = r_old - q;                      // next layer's residual
        qsum[go]  = (layer == 0) ? q : (qsum[go] + q);
    }
    #pragma unroll
    for (int off = 16; off > 0; off >>= 1) sse += __shfl_xor(sse, off, 32);
    if (lane == 0) atomicAdd(&lds_sse, sse);
    __syncthreads();
    if (tid == 0) atomicAdd(sse_accum, lds_sse);

    // ---- indices (stored as float; argmin index per token)
    if (tid < TOK_PER_WG) idx_out[wg_tok0 + tid] = (float)lds_idx[tid];
}

// ---------------------------------------------------------------------------
// loss = sum_l (1 + 0.25) * SSE_l / (N*D)   (q_latent == e_latent numerically)
// ---------------------------------------------------------------------------
__global__ void rvq_finalize(const float* __restrict__ sse, float* __restrict__ loss)
{
    if (threadIdx.x == 0 && blockIdx.x == 0) {
        float s = 0.0f;
        for (int l = 0; l < NUM_LAYERS; ++l) s += sse[l];
        loss[0] = 1.25f * s / (float)((size_t)N_TOKENS * CODE_DIM);
    }
}

// ---------------------------------------------------------------------------
extern "C" void kernel_launch(void* const* d_in, const int* in_sizes, int n_in,
                              void* d_out, int out_size, void* d_ws, size_t ws_size,
                              hipStream_t stream)
{
    const float* z_flat = (const float*)d_in[0];                    // [N, 256]
    const float* cb     = (const float*)d_in[1];                    // [4,1024,256]

    float* out   = (float*)d_out;
    float* loss  = out;                                             // [1]
    float* qsum  = out + 1;                                         // [N,256]
    float* idx0  = out + 1 + (size_t)N_TOKENS * CODE_DIM;           // [4,N]

    // workspace layout
    float*  sse   = (float*)d_ws;                                   // 4 floats
    float*  enorm = sse + 16;                                       // 4096 floats
    __bf16* cb16  = (__bf16*)(enorm + NUM_LAYERS * CB_SIZE);        // 2 MB
    float*  resid = (float*)((char*)d_ws + 16448 + 2097152);        // [N,256] f32

    // 1) prep: bf16 codebooks + code norms + zero SSE accumulators
    rvq_prep<<<(NUM_LAYERS * CB_SIZE + 255) / 256, 256, 0, stream>>>(
        cb, cb16, enorm, sse);

    // 2) four sequential quantization layers
    const int nblocks = N_TOKENS / TOK_PER_WG;   // 1024
    for (int l = 0; l < NUM_LAYERS; ++l) {
        rvq_layer<<<nblocks, WG_THREADS, 0, stream>>>(
            z_flat, resid,
            cb16 + (size_t)l * CB_SIZE * CODE_DIM,
            cb   + (size_t)l * CB_SIZE * CODE_DIM,
            enorm + (size_t)l * CB_SIZE,
            qsum,
            idx0 + (size_t)l * N_TOKENS,
            sse + l,
            l);
    }

    // 3) scalar loss
    rvq_finalize<<<1, 32, 0, stream>>>(sse, loss);
}
